// InterStockAttention_3049426780425
// MI455X (gfx1250) — matrix operand order repaired and checked by reading, not compiled, not run on hardware
//
#include <hip/hip_runtime.h>
#include <hip/hip_bf16.h>

typedef _Float16 f16;
typedef __attribute__((ext_vector_type(16))) _Float16 v16h;
typedef __attribute__((ext_vector_type(8)))  _Float16 v8h;
typedef __attribute__((ext_vector_type(8)))  float    v8f;
typedef __attribute__((ext_vector_type(4)))  float    v4f;
typedef __attribute__((ext_vector_type(4)))  unsigned int u32x4;
typedef __attribute__((ext_vector_type(8)))  int      i32x8;
typedef __attribute__((ext_vector_type(4)))  int      i32x4;

#define S_DIM   512
#define TAU_DIM 512
#define D_DIM   64

// ---- LDS layout (bytes) ----
// sX   : float [512][68]                                       139264
// sK   : f16   [512][64]   (phase3: W1[256][64]+W2[64][256])    65536
// sVt  : f16   [64][528]   (phase3: 8x hbuf[16][264])           67584
// wqo  : f16   Wqkv[192][64] + Wo[64][64]                       32768
// stg  : f16   8x [16][64]                                      16384
// bias : float [832]                                             3328
#define O_SX   0
#define O_SK   139264
#define O_SVT  204800
#define O_WQO  272384
#define O_STG  305152
#define O_BIAS 321536
#define SMEM_BYTES 324864

__device__ __forceinline__ v8f wmma16(v16h a, v16h b, v8f c) {
  return __builtin_amdgcn_wmma_f32_16x16x32_f16(false, a, false, b, (short)0, c,
                                                false, false);
}

// A fragment (16x32 f16, M x K) from f32 row-major buffer, ld in floats.
__device__ __forceinline__ v16h loadA_f32(const float* rowbase, int ld, int col0) {
  int lane = threadIdx.x & 31;
  int m = lane & 15, hf = lane >> 4;
  const float* p = rowbase + m * ld + col0 + 8 * hf;
  v4f a = *(const v4f*)p;
  v4f b = *(const v4f*)(p + 4);
  v4f c = *(const v4f*)(p + 16);
  v4f d = *(const v4f*)(p + 20);
  v16h r;
#pragma unroll
  for (int i = 0; i < 4; ++i) {
    r[i]      = (f16)a[i];
    r[4 + i]  = (f16)b[i];
    r[8 + i]  = (f16)c[i];
    r[12 + i] = (f16)d[i];
  }
  return r;
}

// A fragment (16x32 f16) from f16 row-major buffer, ld in halfs (mult of 8).
__device__ __forceinline__ v16h loadA_h(const f16* base, int ld, int col0) {
  int lane = threadIdx.x & 31;
  int m = lane & 15, hf = lane >> 4;
  const f16* p = base + m * ld + col0 + 8 * hf;
  v8h lo = *(const v8h*)p;
  v8h hi = *(const v8h*)(p + 16);
  v16h r;
#pragma unroll
  for (int i = 0; i < 8; ++i) { r[i] = lo[i]; r[8 + i] = hi[i]; }
  return r;
}

// B fragment (32x16 f16, K x N) from "N-major" buffer: element (k,n) at
// base[(n0+n)*ld + k0 + k]. ld in halfs (mult of 8), k0 mult of 16.
__device__ __forceinline__ v16h loadB_h(const f16* base, int ld, int n0, int k0) {
  int lane = threadIdx.x & 31;
  int n = lane & 15, hf = lane >> 4;
  const f16* p = base + (n0 + n) * ld + k0 + 8 * hf;
  v8h lo = *(const v8h*)p;
  v8h hi = *(const v8h*)(p + 16);
  v16h r;
#pragma unroll
  for (int i = 0; i < 8; ++i) { r[i] = lo[i]; r[8 + i] = hi[i]; }
  return r;
}

__device__ __forceinline__ float redsum16(float x) {
  x += __shfl_xor(x, 1, 32);
  x += __shfl_xor(x, 2, 32);
  x += __shfl_xor(x, 4, 32);
  x += __shfl_xor(x, 8, 32);
  return x;
}
__device__ __forceinline__ float redmax16(float x) {
  x = fmaxf(x, __shfl_xor(x, 1, 32));
  x = fmaxf(x, __shfl_xor(x, 2, 32));
  x = fmaxf(x, __shfl_xor(x, 4, 32));
  x = fmaxf(x, __shfl_xor(x, 8, 32));
  return x;
}

__device__ __forceinline__ void lds_fence() { asm volatile("" ::: "memory"); }

__global__ __launch_bounds__(256) void interstock_attn_layer(
    const float* __restrict__ x, const float* __restrict__ Wqkv,
    const float* __restrict__ bqkv, const float* __restrict__ Wo,
    const float* __restrict__ bo, const float* __restrict__ g1,
    const float* __restrict__ be1, const float* __restrict__ W1,
    const float* __restrict__ bf1, const float* __restrict__ W2,
    const float* __restrict__ bf2, const float* __restrict__ g2,
    const float* __restrict__ be2, float* __restrict__ out) {
  __shared__ __align__(16) unsigned char smem[SMEM_BYTES];
  float* sX  = (float*)(smem + O_SX);    // [512][68]
  f16*   sK  = (f16*)(smem + O_SK);      // [512][64]
  f16*   sVt = (f16*)(smem + O_SVT);     // [64][528]
  f16*   sWq = (f16*)(smem + O_WQO);     // [192][64]
  f16*   sWo = sWq + 192 * 64;           // [64][64]
  f16*   sStg = (f16*)(smem + O_STG);    // 8 x [16][64]
  float* sB  = (float*)(smem + O_BIAS);  // bqkv0 bo192 bf1:256 bf2:512 g1:576 be1:640 g2:704 be2:768

  const int tid  = threadIdx.x;
  const int t    = blockIdx.x;
  const int wave = tid >> 5;
  const int lane = tid & 31;
  const int ln   = lane & 15;   // N index within D-tile
  const int lh   = lane >> 4;   // lane half

  // ---------------- Phase 0: TDM-load x slice; stage weights/biases --------
  if (wave == 0) {
    // Tensor DMA: 2D tile 512 rows x 64 f32, row stride tau*D, into sX with
    // 4-DWORD padding every 64 DWORDs (-> [512][68] layout).
    unsigned long long ga =
        (unsigned long long)(uintptr_t)x + (unsigned long long)t * (D_DIM * 4);
    unsigned int ldsa = (unsigned int)(uintptr_t)(void*)sX;
    u32x4 g0;
    g0[0] = 1u;                                   // count=1 (valid), user mode
    g0[1] = ldsa;                                 // lds_addr (bytes)
    g0[2] = (unsigned int)ga;                     // global_addr[31:0]
    g0[3] = (unsigned int)((ga >> 32) & 0x1FFFFFFu) | 0x80000000u;  // [56:32] | type=2
    i32x8 g1v;
    g1v[0] = (2 << 16) | (1 << 20) | (5 << 22) | (3 << 25);
    //        data=4B    pad_en      ivl=64dw     amt=4dw
    g1v[1] = (64 << 16);   // tensor_dim0 = 64          (bits 79:48)
    g1v[2] = (512 << 16);  // tensor_dim1 = 512         (bits 111:80)
    g1v[3] = (64 << 16);   // tile_dim0 = 64            (bits 127:112)
    g1v[4] = 512;          // tile_dim1 = 512, tile_dim2 = 0
    g1v[5] = 32768;        // tensor_dim0_stride = tau*D (bits 207:160)
    g1v[6] = 0;
    g1v[7] = 0;
    i32x4 gz = {0, 0, 0, 0};
    i32x8 gz8 = {0, 0, 0, 0, 0, 0, 0, 0};
    __builtin_amdgcn_tensor_load_to_lds(g0, g1v, gz, gz, gz8, 0);
  }
  // Warm L2 for phase-3 FFN weights while the TDM runs.
  for (int i = tid; i < 512; i += 256) {
    __builtin_prefetch(W1 + i * 32, 0, 3);
    __builtin_prefetch(W2 + i * 32, 0, 3);
  }
  for (int i = tid; i < 192 * 64; i += 256) sWq[i] = (f16)Wqkv[i];
  for (int i = tid; i < 64 * 64; i += 256) sWo[i] = (f16)Wo[i];
  for (int i = tid; i < 192; i += 256) sB[i] = bqkv[i];
  for (int i = tid; i < 256; i += 256) sB[256 + i] = bf1[i];
  for (int i = tid; i < 64; i += 256) {
    sB[192 + i] = bo[i];
    sB[512 + i] = bf2[i];
    sB[576 + i] = g1[i];
    sB[640 + i] = be1[i];
    sB[704 + i] = g2[i];
    sB[768 + i] = be2[i];
  }
  if (wave == 0) __builtin_amdgcn_s_wait_tensorcnt(0);
  __syncthreads();

  // ---------------- Phase 1: K, V for all stocks ---------------------------
  for (int it = 0; it < 4; ++it) {
    int m0 = (wave + it * 8) * 16;
    v16h a0 = loadA_f32(sX + m0 * 68, 68, 0);
    v16h a1 = loadA_f32(sX + m0 * 68, 68, 32);
#pragma unroll
    for (int nt = 0; nt < 8; ++nt) {
      int n0 = 64 + nt * 16;  // Wqkv rows 64..191 (K then V)
      v8f acc = {};
      acc = wmma16(a0, loadB_h(sWq, 64, n0, 0), acc);
      acc = wmma16(a1, loadB_h(sWq, 64, n0, 32), acc);
      float b = sB[n0 + ln];
      if (nt < 4) {
        int c = n0 - 64 + ln;
#pragma unroll
        for (int v = 0; v < 8; ++v)
          sK[(m0 + 8 * lh + v) * 64 + c] = (f16)(acc[v] + b);
      } else {
        int c = n0 - 128 + ln;
#pragma unroll
        for (int v = 0; v < 8; ++v)
          sVt[c * 528 + (m0 + 8 * lh + v)] = (f16)(acc[v] + b);
      }
    }
  }
  __syncthreads();

  // ---------------- Phase 2: attention + Wo + residual + LN1 ---------------
  const float scale = 0.17677669529663687f;  // 1/sqrt(32)
  f16* stg = sStg + wave * 16 * 64;
  for (int it = 0; it < 4; ++it) {
    int m0 = (wave + it * 8) * 16;
    v16h a0 = loadA_f32(sX + m0 * 68, 68, 0);
    v16h a1 = loadA_f32(sX + m0 * 68, 68, 32);
    // q tile (16x64) -> stg
#pragma unroll
    for (int nt = 0; nt < 4; ++nt) {
      int n0 = nt * 16;
      v8f acc = {};
      acc = wmma16(a0, loadB_h(sWq, 64, n0, 0), acc);
      acc = wmma16(a1, loadB_h(sWq, 64, n0, 32), acc);
      float b = sB[n0 + ln];
#pragma unroll
      for (int v = 0; v < 8; ++v)
        stg[(8 * lh + v) * 64 + n0 + ln] = (f16)(acc[v] + b);
    }
    lds_fence();
    v16h qA0 = loadA_h(stg, 64, 0);
    v16h qA1 = loadA_h(stg, 64, 32);
    lds_fence();

    for (int h = 0; h < 2; ++h) {
      v16h qA = h ? qA1 : qA0;
      float mrun[8], lrun[8];
#pragma unroll
      for (int v = 0; v < 8; ++v) { mrun[v] = -1e30f; lrun[v] = 0.f; }
      v8f c0 = {}, c1 = {};
      for (int jc = 0; jc < 16; ++jc) {
        int j0 = jc * 32;
        v8f z = {};
        v8f s0 = wmma16(qA, loadB_h(sK, 64, j0, h * 32), z);
        v8f s1 = wmma16(qA, loadB_h(sK, 64, j0 + 16, h * 32), z);
#pragma unroll
        for (int v = 0; v < 8; ++v) {
          float sa = s0[v] * scale, sb = s1[v] * scale;
          float mx = redmax16(fmaxf(sa, sb));
          float nm = fmaxf(mrun[v], mx);
          float pa = __expf(sa - nm), pb = __expf(sb - nm);
          float corr = __expf(mrun[v] - nm);
          float rs = redsum16(pa + pb);
          lrun[v] = lrun[v] * corr + rs;
          mrun[v] = nm;
          c0[v] *= corr;
          c1[v] *= corr;
          stg[(8 * lh + v) * 64 + 32 + ln] = (f16)pa;
          stg[(8 * lh + v) * 64 + 48 + ln] = (f16)pb;
        }
        lds_fence();
        v16h pA = loadA_h(stg, 64, 32);  // P: 16 x 32 keys
        c0 = wmma16(pA, loadB_h(sVt, 528, h * 32, j0), c0);
        c1 = wmma16(pA, loadB_h(sVt, 528, h * 32 + 16, j0), c1);
        lds_fence();
      }
#pragma unroll
      for (int v = 0; v < 8; ++v) {
        float inv = 1.f / lrun[v];
        stg[(8 * lh + v) * 64 + h * 32 + ln]      = (f16)(c0[v] * inv);
        stg[(8 * lh + v) * 64 + h * 32 + 16 + ln] = (f16)(c1[v] * inv);
      }
      lds_fence();
    }
    v16h cA0 = loadA_h(stg, 64, 0);
    v16h cA1 = loadA_h(stg, 64, 32);

    // attn_out = ctx @ Wo^T + bo; residual + LN1 -> overwrite sX rows
    v8f zt[4];
    float rs[8], rq[8];
#pragma unroll
    for (int v = 0; v < 8; ++v) { rs[v] = 0.f; rq[v] = 0.f; }
#pragma unroll
    for (int nt = 0; nt < 4; ++nt) {
      int n0 = nt * 16;
      v8f acc = {};
      acc = wmma16(cA0, loadB_h(sWo, 64, n0, 0), acc);
      acc = wmma16(cA1, loadB_h(sWo, 64, n0, 32), acc);
      float b = sB[192 + n0 + ln];
#pragma unroll
      for (int v = 0; v < 8; ++v) {
        float val = acc[v] + b + sX[(m0 + 8 * lh + v) * 68 + n0 + ln];
        zt[nt][v] = val;
        rs[v] += val;
        rq[v] += val * val;
      }
    }
    float mu[8], rinv[8];
#pragma unroll
    for (int v = 0; v < 8; ++v) {
      float s = redsum16(rs[v]);
      float q = redsum16(rq[v]);
      mu[v] = s * (1.f / 64.f);
      float var = q * (1.f / 64.f) - mu[v] * mu[v];
      rinv[v] = rsqrtf(var + 1e-5f);
    }
#pragma unroll
    for (int nt = 0; nt < 4; ++nt) {
      int n0 = nt * 16;
      float g = sB[576 + n0 + ln], be = sB[640 + n0 + ln];
#pragma unroll
      for (int v = 0; v < 8; ++v) {
        float y = (zt[nt][v] - mu[v]) * rinv[v] * g + be;
        sX[(m0 + 8 * lh + v) * 68 + n0 + ln] = y;  // own rows only
      }
    }
  }
  __syncthreads();

  // ---------------- Phase 3: FFN + residual + LN2 --------------------------
  f16* w1 = sK;             // [256][64]
  f16* w2 = sK + 256 * 64;  // [64][256]
  for (int i = tid; i < 256 * 64; i += 256) w1[i] = (f16)W1[i];
  for (int i = tid; i < 64 * 256; i += 256) w2[i] = (f16)W2[i];
  __syncthreads();
  f16* hb = sVt + wave * (16 * 264);  // per-wave hidden [16][264]

  for (int it = 0; it < 4; ++it) {
    int m0 = (wave + it * 8) * 16;
    v16h a0 = loadA_f32(sX + m0 * 68, 68, 0);
    v16h a1 = loadA_f32(sX + m0 * 68, 68, 32);
#pragma unroll
    for (int nt = 0; nt < 16; ++nt) {
      int n0 = nt * 16;
      v8f acc = {};
      acc = wmma16(a0, loadB_h(w1, 64, n0, 0), acc);
      acc = wmma16(a1, loadB_h(w1, 64, n0, 32), acc);
      float b = sB[256 + n0 + ln];
#pragma unroll
      for (int v = 0; v < 8; ++v)
        hb[(8 * lh + v) * 264 + n0 + ln] = (f16)fmaxf(acc[v] + b, 0.f);
    }
    lds_fence();

    v8f zt[4];
    float rs[8], rq[8];
#pragma unroll
    for (int v = 0; v < 8; ++v) { rs[v] = 0.f; rq[v] = 0.f; }
#pragma unroll
    for (int nt = 0; nt < 4; ++nt) {
      int n0 = nt * 16;
      v8f acc = {};
#pragma unroll
      for (int kk = 0; kk < 8; ++kk)
        acc = wmma16(loadA_h(hb, 264, kk * 32), loadB_h(w2, 256, n0, kk * 32), acc);
      float b = sB[512 + n0 + ln];
#pragma unroll
      for (int v = 0; v < 8; ++v) {
        float val = acc[v] + b + sX[(m0 + 8 * lh + v) * 68 + n0 + ln];
        zt[nt][v] = val;
        rs[v] += val;
        rq[v] += val * val;
      }
    }
    float mu[8], rinv[8];
#pragma unroll
    for (int v = 0; v < 8; ++v) {
      float s = redsum16(rs[v]);
      float q = redsum16(rq[v]);
      mu[v] = s * (1.f / 64.f);
      float var = q * (1.f / 64.f) - mu[v] * mu[v];
      rinv[v] = rsqrtf(var + 1e-5f);
    }
#pragma unroll
    for (int nt = 0; nt < 4; ++nt) {
      int n0 = nt * 16;
      float g = sB[704 + n0 + ln], be = sB[768 + n0 + ln];
#pragma unroll
      for (int v = 0; v < 8; ++v) {
        float zv = (zt[nt][v] - mu[v]) * rinv[v] * g + be;
        size_t srow = (size_t)(m0 + 8 * lh + v);
        out[srow * (TAU_DIM * D_DIM) + (size_t)t * D_DIM + n0 + ln] = zv;
      }
    }
    lds_fence();
  }
}

extern "C" void kernel_launch(void* const* d_in, const int* in_sizes, int n_in,
                              void* d_out, int out_size, void* d_ws, size_t ws_size,
                              hipStream_t stream) {
  (void)in_sizes; (void)n_in; (void)out_size; (void)d_ws; (void)ws_size;
  const float* x    = (const float*)d_in[0];
  const float* Wqkv = (const float*)d_in[1];
  const float* bqkv = (const float*)d_in[2];
  const float* Wo   = (const float*)d_in[3];
  const float* bo   = (const float*)d_in[4];
  const float* g1   = (const float*)d_in[5];
  const float* be1  = (const float*)d_in[6];
  const float* W1   = (const float*)d_in[7];
  const float* bf1  = (const float*)d_in[8];
  const float* W2   = (const float*)d_in[9];
  const float* bf2  = (const float*)d_in[10];
  const float* g2   = (const float*)d_in[11];
  const float* be2  = (const float*)d_in[12];
  float* out = (float*)d_out;

  interstock_attn_layer<<<dim3(TAU_DIM), dim3(256), 0, stream>>>(
      x, Wqkv, bqkv, Wo, bo, g1, be1, W1, bf1, W2, bf2, g2, be2, out);
}
